// MultiHeadAttention_47682726920291
// MI455X (gfx1250) — matrix operand order repaired and checked by reading, not compiled, not run on hardware
//
#include <hip/hip_runtime.h>
#include <hip/hip_bf16.h>

typedef _Float16 half_t;
typedef __attribute__((ext_vector_type(16))) _Float16     v16h;
typedef __attribute__((ext_vector_type(8)))  float        v8f;
typedef __attribute__((ext_vector_type(8)))  unsigned int v8u;
typedef __attribute__((ext_vector_type(4)))  int          v4i;

#define B_      4
#define L_      2048
#define DIM_    512
#define HEADS_  8
#define HD_     64
#define MTOT_   (B_ * L_)            // 8192 token rows
#define SCALE_  0.125f               // 64^-0.5
#define NEGV_   (-1e9f)
#define RPE_W_  (2 * 2048 - 1)       // 4095
#define RPE_C_  2047                 // RPE_SIZE - 1

#if defined(__has_builtin)
#if __has_builtin(__builtin_amdgcn_global_load_async_to_lds_b128) && \
    __has_builtin(__builtin_amdgcn_s_wait_asynccnt)
#define USE_ASYNC_LDS 1
#endif
#endif
#ifndef USE_ASYNC_LDS
#define USE_ASYNC_LDS 0
#endif

static __device__ __forceinline__ void wait_ds0() {
#if __has_builtin(__builtin_amdgcn_s_wait_dscnt)
  __builtin_amdgcn_s_wait_dscnt(0);
#else
  asm volatile("s_wait_dscnt 0x0" ::: "memory");
#endif
}

static __device__ __forceinline__ void wait_async0() {
#if USE_ASYNC_LDS
  __builtin_amdgcn_s_wait_asynccnt(0);
#endif
}

// copy 16 bytes global -> LDS (async-to-LDS on CDNA5; bypasses VGPRs)
static __device__ __forceinline__ void copy16_g2l(const half_t* g, half_t* l) {
#if USE_ASYNC_LDS
  __builtin_amdgcn_global_load_async_to_lds_b128(
      (__attribute__((address_space(1))) v4i*)g,
      (__attribute__((address_space(3))) v4i*)l, 0, 0);
#else
  *(float4*)l = *(const float4*)g;
#endif
}

static __device__ __forceinline__ v8f wmma_f16(v16h a, v16h b, v8f c) {
  return __builtin_amdgcn_wmma_f32_16x16x32_f16(false, a, false, b, (short)0, c,
                                                false, false);
}

// ---------------------------------------------------------------------------
// f32 -> f16 convert, 4 elements / thread (n must be a multiple of 4)
// ---------------------------------------------------------------------------
__global__ void __launch_bounds__(256)
cvt_f32_f16_kernel(const float* __restrict__ src, half_t* __restrict__ dst, int n4) {
  int i = blockIdx.x * blockDim.x + threadIdx.x;
  if (i >= n4) return;
  float4 f = ((const float4*)src)[i];
  union { _Float16 h[4]; uint2 u; } p;
  p.h[0] = (half_t)f.x; p.h[1] = (half_t)f.y;
  p.h[2] = (half_t)f.z; p.h[3] = (half_t)f.w;
  ((uint2*)dst)[i] = p.u;
}

// ---------------------------------------------------------------------------
// Y = X (8192x512, f16) @ W^T (W is 512x512 row-major [n][k], f16) + bias(f32)
// Block: 256 threads (8 waves), 128x128 output tile, K-steps of 32,
// async-to-LDS double-buffered staging. Each wave: 16x128 strip, 8 WMMAs/step.
// MODE 0: Q  -> (val*SCALE) f16 at [B,H,L,64]
// MODE 1: K  ->  val        f16 at [B,H,L,64]
// MODE 2: V  ->  val        f16 at [B,H,64,L]   (transposed for P.V B-frags)
// MODE 3: out->  val        f32 at [B,L,512], zeroed where q_padding_mask
// ---------------------------------------------------------------------------
template <int MODE>
__global__ void __launch_bounds__(256)
gemm512_kernel(const half_t* __restrict__ X, const half_t* __restrict__ W,
               const float* __restrict__ bias, half_t* __restrict__ outH,
               float* __restrict__ outF, const unsigned char* __restrict__ qmask) {
  __shared__ half_t As[2][128 * 32];   // 2 x 8 KB
  __shared__ half_t Bs[2][128 * 32];   // 2 x 8 KB

  const int tid   = threadIdx.x;
  const int wave  = tid >> 5;
  const int lane  = tid & 31;
  const int lmod  = lane & 15;
  const int lhalf = lane >> 4;
  const int mBase = blockIdx.x * 128;
  const int nBase = blockIdx.y * 128;

  // per-thread staging coordinates: 128 rows x 32 halves, 2 threads/row, 32B each
  const int srow = tid >> 1;
  const int shh  = (tid & 1) * 16;
  const half_t* aSrc = X + (size_t)(mBase + srow) * DIM_ + shh;
  const half_t* bSrc = W + (size_t)(nBase + srow) * DIM_ + shh;

  auto stage = [&](int buf, int k0) {
    copy16_g2l(aSrc + k0,     &As[buf][srow * 32 + shh]);
    copy16_g2l(aSrc + k0 + 8, &As[buf][srow * 32 + shh + 8]);
    copy16_g2l(bSrc + k0,     &Bs[buf][srow * 32 + shh]);
    copy16_g2l(bSrc + k0 + 8, &Bs[buf][srow * 32 + shh + 8]);
  };

  v8f acc[8] = {};

  stage(0, 0);
  wait_async0();
  __syncthreads();

  for (int it = 0; it < DIM_ / 32; ++it) {
    const int cur = it & 1;
    const int k0n = (it + 1) * 32;
    if (k0n < DIM_) stage(cur ^ 1, k0n);   // prefetch next K-slice (other buffer)

    // A fragment: row = wave*16 + lmod; K = lhalf*16 + e (contiguous, matches B)
    v16h afrag;
    {
      const unsigned int* ar =
          (const unsigned int*)(&As[cur][(wave * 16 + lmod) * 32]);
      v8u t;
#pragma unroll
      for (int p = 0; p < 4; ++p) t[p] = ar[lhalf * 8 + p];
#pragma unroll
      for (int p = 0; p < 4; ++p) t[4 + p] = ar[lhalf * 8 + 4 + p];
      afrag = __builtin_bit_cast(v16h, t);
    }
#pragma unroll
    for (int nt = 0; nt < 8; ++nt) {
      // B fragment: n = nt*16 + lmod, K = lhalf*16 + e  (contiguous in Bs row)
      const unsigned int* br =
          (const unsigned int*)(&Bs[cur][(nt * 16 + lmod) * 32]);
      v8u t;
#pragma unroll
      for (int p = 0; p < 8; ++p) t[p] = br[lhalf * 8 + p];
      v16h bfrag = __builtin_bit_cast(v16h, t);
      acc[nt] = wmma_f16(afrag, bfrag, acc[nt]);
    }

    wait_async0();      // our async writes into buf cur^1 are done
    __syncthreads();    // everyone done reading buf cur / writing buf cur^1
  }

  // epilogue: C layout row m = r + 8*lhalf, col n = lmod
#pragma unroll
  for (int nt = 0; nt < 8; ++nt) {
    const int nG = nBase + nt * 16 + lmod;
    const float bv = bias[nG];
    const int h = nG >> 6, d = nG & 63;
#pragma unroll
    for (int r = 0; r < 8; ++r) {
      const int mG = mBase + wave * 16 + r + 8 * lhalf;
      const int b = mG >> 11, l = mG & (L_ - 1);
      float val = acc[nt][r] + bv;
      if (MODE == 0) {
        outH[(((size_t)(b * HEADS_ + h) * L_) + l) * HD_ + d] = (half_t)(val * SCALE_);
      } else if (MODE == 1) {
        outH[(((size_t)(b * HEADS_ + h) * L_) + l) * HD_ + d] = (half_t)val;
      } else if (MODE == 2) {
        outH[(((size_t)(b * HEADS_ + h) * HD_) + d) * L_ + l] = (half_t)val;
      } else {
        if (qmask[mG]) val = 0.0f;
        outF[(size_t)mG * DIM_ + nG] = val;
      }
    }
  }
}

// ---------------------------------------------------------------------------
// Flash attention. Block = 256 threads = 8 independent waves; each wave owns a
// 16-query tile of one (b,h). S = Q.K^T (scale folded into Q), + rpe bias,
// key-mask to -1e9, online softmax, O += P.V via LDS-transposed P.
// ---------------------------------------------------------------------------
__global__ void __launch_bounds__(256)
flash_attn_kernel(const half_t* __restrict__ Qp,   // [B,H,L,64] (pre-scaled)
                  const half_t* __restrict__ Kp,   // [B,H,L,64]
                  const half_t* __restrict__ Vt,   // [B,H,64,L]
                  const float* __restrict__ rpe,   // [8,4095]
                  const unsigned char* __restrict__ kmask,  // [B,L]
                  half_t* __restrict__ O) {        // [B,L,512] f16
  __shared__ half_t Pl[8 * 16 * 32];  // per-wave private 16x32 P tile, 8 KB

  const int tid = threadIdx.x, wave = tid >> 5, lane = tid & 31;
  const int lmod = lane & 15, lhalf = lane >> 4;
  const int bh   = blockIdx.x / (L_ / 128);
  const int qblk = blockIdx.x % (L_ / 128);
  const int b = bh / HEADS_, h = bh % HEADS_;
  const int qbase = qblk * 128 + wave * 16;

  const half_t* Qb = Qp + ((size_t)bh * L_ + qbase) * HD_;
  const half_t* Kb = Kp + (size_t)bh * L_ * HD_;
  const half_t* Vb = Vt + (size_t)bh * HD_ * L_;
  const float*  rh = rpe + h * RPE_W_;
  const unsigned char* km = kmask + b * L_;

  // Q A-fragments for dim slices [0,32) and [32,64)
  v16h qa[2];
  {
    const unsigned int* qr = (const unsigned int*)(Qb + lmod * HD_);
#pragma unroll
    for (int ks = 0; ks < 2; ++ks) {
      v8u t;
#pragma unroll
      for (int p = 0; p < 4; ++p) t[p] = qr[ks * 16 + lhalf * 8 + p];
#pragma unroll
      for (int p = 0; p < 4; ++p) t[4 + p] = qr[ks * 16 + lhalf * 8 + 4 + p];
      qa[ks] = __builtin_bit_cast(v16h, t);
    }
  }

  v8f acc[4] = {};
  float mrun[8], lrun[8];
#pragma unroll
  for (int r = 0; r < 8; ++r) { mrun[r] = -1e30f; lrun[r] = 0.0f; }

  half_t* pl = Pl + wave * 16 * 32;

  for (int kb = 0; kb < L_; kb += 32) {
    // prefetch next key/value chunk into cache (global_prefetch_b8)
    if (kb + 32 < L_) {
      __builtin_prefetch(Kb + (size_t)(kb + 32 + lane) * HD_, 0, 1);
      __builtin_prefetch(Vb + (size_t)lane * L_ + kb + 32, 0, 1);
    }

    // ---- S = Q.K^T for 16x32 score tile (two 16x16 C fragments) ----
    v8f S[2];
#pragma unroll
    for (int jt = 0; jt < 2; ++jt) {
      v8f s = {};
#pragma unroll
      for (int ks = 0; ks < 2; ++ks) {
        const unsigned int* kr =
            (const unsigned int*)(Kb + (size_t)(kb + jt * 16 + lmod) * HD_);
        v8u t;
#pragma unroll
        for (int p = 0; p < 8; ++p) t[p] = kr[ks * 16 + lhalf * 8 + p];
        s = wmma_f16(qa[ks], __builtin_bit_cast(v16h, t), s);
      }
      S[jt] = s;
    }

    // ---- rpe bias + key padding mask ----
#pragma unroll
    for (int jt = 0; jt < 2; ++jt) {
      const int j = kb + jt * 16 + lmod;
      const bool masked = km[j] != 0;
#pragma unroll
      for (int r = 0; r < 8; ++r) {
        const int i = qbase + r + 8 * lhalf;
        const float bias = rh[j - i + RPE_C_];
        S[jt][r] = masked ? NEGV_ : (S[jt][r] + bias);
      }
    }

    // ---- online softmax (row stats replicated across each 16-lane half) ----
    float P0[8], P1[8], resc[8];
#pragma unroll
    for (int r = 0; r < 8; ++r) {
      float mx = fmaxf(S[0][r], S[1][r]);
#pragma unroll
      for (int off = 1; off < 16; off <<= 1) mx = fmaxf(mx, __shfl_xor(mx, off, 16));
      const float mnew = fmaxf(mrun[r], mx);
      resc[r] = __expf(mrun[r] - mnew);
      P0[r] = __expf(S[0][r] - mnew);
      P1[r] = __expf(S[1][r] - mnew);
      float sum = P0[r] + P1[r];
#pragma unroll
      for (int off = 1; off < 16; off <<= 1) sum += __shfl_xor(sum, off, 16);
      lrun[r] = lrun[r] * resc[r] + sum;
      mrun[r] = mnew;
    }
#pragma unroll
    for (int nt = 0; nt < 4; ++nt)
#pragma unroll
      for (int r = 0; r < 8; ++r) acc[nt][r] *= resc[r];

    // ---- C-layout P -> LDS -> A-layout fragment ----
#pragma unroll
    for (int r = 0; r < 8; ++r) {
      const int m = r + 8 * lhalf;
      pl[m * 32 + lmod]      = (half_t)P0[r];
      pl[m * 32 + 16 + lmod] = (half_t)P1[r];
    }
    wait_ds0();
    v16h pa;
    {
      const unsigned int* pr = (const unsigned int*)(pl + lmod * 32);
      v8u t;
#pragma unroll
      for (int p = 0; p < 4; ++p) t[p] = pr[lhalf * 8 + p];
#pragma unroll
      for (int p = 0; p < 4; ++p) t[4 + p] = pr[lhalf * 8 + 4 + p];
      pa = __builtin_bit_cast(v16h, t);
    }

    // ---- O += P.V  (V transposed: B-frag rows contiguous along keys) ----
#pragma unroll
    for (int nt = 0; nt < 4; ++nt) {
      const unsigned int* vr =
          (const unsigned int*)(Vb + (size_t)(nt * 16 + lmod) * L_ + kb);
      v8u t;
#pragma unroll
      for (int p = 0; p < 8; ++p) t[p] = vr[lhalf * 8 + p];
      acc[nt] = wmma_f16(pa, __builtin_bit_cast(v16h, t), acc[nt]);
    }
  }

  // ---- normalize + store O (f16, [B,L,512]) ----
  float inv[8];
#pragma unroll
  for (int r = 0; r < 8; ++r) inv[r] = 1.0f / lrun[r];
#pragma unroll
  for (int nt = 0; nt < 4; ++nt)
#pragma unroll
    for (int r = 0; r < 8; ++r) {
      const int m = r + 8 * lhalf;
      O[((size_t)(b * L_ + qbase + m)) * DIM_ + h * HD_ + nt * 16 + lmod] =
          (half_t)(acc[nt][r] * inv[r]);
    }
}

// ---------------------------------------------------------------------------
extern "C" void kernel_launch(void* const* d_in, const int* in_sizes, int n_in,
                              void* d_out, int out_size, void* d_ws, size_t ws_size,
                              hipStream_t stream) {
  const float* q   = (const float*)d_in[0];
  const float* k   = (const float*)d_in[1];
  const float* v   = (const float*)d_in[2];
  const float* Wq  = (const float*)d_in[3];
  const float* bq  = (const float*)d_in[4];
  const float* Wk  = (const float*)d_in[5];
  const float* bk  = (const float*)d_in[6];
  const float* Wv  = (const float*)d_in[7];
  const float* bv  = (const float*)d_in[8];
  const float* Wo  = (const float*)d_in[9];
  const float* bo  = (const float*)d_in[10];
  const float* rpe = (const float*)d_in[11];
  const unsigned char* qmask = (const unsigned char*)d_in[12];
  const unsigned char* kmask = (const unsigned char*)d_in[13];
  float* out = (float*)d_out;

  const size_t NX = (size_t)MTOT_ * DIM_;   // 4,194,304
  const size_t NW = (size_t)DIM_ * DIM_;    // 262,144

  char* p = (char*)d_ws;
  auto carve = [&](size_t bytes) {
    void* r = (void*)p;
    p += (bytes + 255) & ~(size_t)255;
    return r;
  };
  half_t* q16  = (half_t*)carve(NX * 2);
  half_t* k16  = (half_t*)carve(NX * 2);
  half_t* v16  = (half_t*)carve(NX * 2);
  half_t* Wq16 = (half_t*)carve(NW * 2);
  half_t* Wk16 = (half_t*)carve(NW * 2);
  half_t* Wv16 = (half_t*)carve(NW * 2);
  half_t* Wo16 = (half_t*)carve(NW * 2);
  half_t* Qp   = (half_t*)carve(NX * 2);    // [B,H,L,64], pre-scaled
  half_t* Kp   = (half_t*)carve(NX * 2);    // [B,H,L,64]
  half_t* Vtp  = (half_t*)carve(NX * 2);    // [B,H,64,L]
  half_t* O16  = (half_t*)carve(NX * 2);    // [B,L,512]
  (void)ws_size; (void)n_in; (void)in_sizes; (void)out_size;

  // 1) f32 -> f16 conversions
  {
    const int nx4 = (int)(NX / 4), nw4 = (int)(NW / 4);
    dim3 bx(256);
    cvt_f32_f16_kernel<<<dim3((nx4 + 255) / 256), bx, 0, stream>>>(q, q16, nx4);
    cvt_f32_f16_kernel<<<dim3((nx4 + 255) / 256), bx, 0, stream>>>(k, k16, nx4);
    cvt_f32_f16_kernel<<<dim3((nx4 + 255) / 256), bx, 0, stream>>>(v, v16, nx4);
    cvt_f32_f16_kernel<<<dim3((nw4 + 255) / 256), bx, 0, stream>>>(Wq, Wq16, nw4);
    cvt_f32_f16_kernel<<<dim3((nw4 + 255) / 256), bx, 0, stream>>>(Wk, Wk16, nw4);
    cvt_f32_f16_kernel<<<dim3((nw4 + 255) / 256), bx, 0, stream>>>(Wv, Wv16, nw4);
    cvt_f32_f16_kernel<<<dim3((nw4 + 255) / 256), bx, 0, stream>>>(Wo, Wo16, nw4);
  }

  // 2) projections (WMMA GEMMs, async-LDS double-buffered)
  {
    dim3 grid(MTOT_ / 128, DIM_ / 128), blk(256);
    gemm512_kernel<0><<<grid, blk, 0, stream>>>(q16, Wq16, bq, Qp, nullptr, nullptr);
    gemm512_kernel<1><<<grid, blk, 0, stream>>>(k16, Wk16, bk, Kp, nullptr, nullptr);
    gemm512_kernel<2><<<grid, blk, 0, stream>>>(v16, Wv16, bv, Vtp, nullptr, nullptr);
  }

  // 3) fused flash attention (+rpe, +key mask, softmax)
  {
    dim3 grid(B_ * HEADS_ * (L_ / 128)), blk(256);
    flash_attn_kernel<<<grid, blk, 0, stream>>>(Qp, Kp, Vtp, rpe, kmask, O16);
  }

  // 4) output projection + bias + query-mask zeroing -> f32 d_out
  {
    dim3 grid(MTOT_ / 128, DIM_ / 128), blk(256);
    gemm512_kernel<3><<<grid, blk, 0, stream>>>(O16, Wo16, bo, nullptr, out, qmask);
  }
}
